// QuantLinear_56530359550878
// MI455X (gfx1250) — 8-bit matrix operand order repaired and checked by reading, compiled, not run on hardware
//
#include <hip/hip_runtime.h>
#include <hip/hip_bf16.h>
#include <stdint.h>

typedef __attribute__((ext_vector_type(8))) int v8i;

#define QMAX_W 127.0f
#define QMAX_A 255.0f

__global__ __launch_bounds__(256) void qw_kernel(const float* __restrict__ W,
                                                 int8_t* __restrict__ Wq,
                                                 float* __restrict__ wscale,
                                                 int* __restrict__ wsum,
                                                 int K) {
    __shared__ float sf[256];
    __shared__ int   si[256];
    const int n = blockIdx.x;
    const int t = threadIdx.x;
    const float* row = W + (size_t)n * K;

    float4 v[4];
    float am = 0.0f;
#pragma unroll
    for (int i = 0; i < 4; ++i) {
        v[i] = *(const float4*)(row + i * 1024 + t * 4);
        am = fmaxf(am, fmaxf(fmaxf(fabsf(v[i].x), fabsf(v[i].y)),
                             fmaxf(fabsf(v[i].z), fabsf(v[i].w))));
    }
    sf[t] = am;
    __syncthreads();
    for (int s = 128; s > 0; s >>= 1) {
        if (t < s) sf[t] = fmaxf(sf[t], sf[t + s]);
        __syncthreads();
    }
    const float absmax = sf[0];
    const float scale  = (absmax > 0.0f) ? (absmax / QMAX_W) : 1.0f;
    const float inv    = 1.0f / scale;

    int qs = 0;
#pragma unroll
    for (int i = 0; i < 4; ++i) {
        int q0 = (int)rintf(v[i].x * inv); q0 = min(127, max(-127, q0));
        int q1 = (int)rintf(v[i].y * inv); q1 = min(127, max(-127, q1));
        int q2 = (int)rintf(v[i].z * inv); q2 = min(127, max(-127, q2));
        int q3 = (int)rintf(v[i].w * inv); q3 = min(127, max(-127, q3));
        qs += q0 + q1 + q2 + q3;
        const int packed = (q0 & 255) | ((q1 & 255) << 8) |
                           ((q2 & 255) << 16) | ((q3 & 255) << 24);
        *(int*)(Wq + (size_t)n * K + i * 1024 + t * 4) = packed;
    }
    si[t] = qs;
    __syncthreads();
    for (int s = 128; s > 0; s >>= 1) {
        if (t < s) si[t] += si[t + s];
        __syncthreads();
    }
    if (t == 0) {
        wscale[n] = scale;
        wsum[n]   = si[0];
    }
}

__global__ __launch_bounds__(256) void qx_kernel(const float* __restrict__ X,
                                                 uint8_t* __restrict__ Xq,
                                                 float* __restrict__ ascale,
                                                 float* __restrict__ azp,
                                                 int K) {
    __shared__ float smin[256];
    __shared__ float smax[256];
    const int m = blockIdx.x;
    const int t = threadIdx.x;
    const float* row = X + (size_t)m * K;

    float4 v[4];
    float lmin =  3.4e38f;
    float lmax = -3.4e38f;
#pragma unroll
    for (int i = 0; i < 4; ++i) {
        v[i] = *(const float4*)(row + i * 1024 + t * 4);
        lmin = fminf(lmin, fminf(fminf(v[i].x, v[i].y), fminf(v[i].z, v[i].w)));
        lmax = fmaxf(lmax, fmaxf(fmaxf(v[i].x, v[i].y), fmaxf(v[i].z, v[i].w)));
    }
    smin[t] = lmin;
    smax[t] = lmax;
    __syncthreads();
    for (int s = 128; s > 0; s >>= 1) {
        if (t < s) {
            smin[t] = fminf(smin[t], smin[t + s]);
            smax[t] = fmaxf(smax[t], smax[t + s]);
        }
        __syncthreads();
    }
    const float xmin  = smin[0];
    const float rng   = smax[0] - xmin;
    const float scale = (rng > 0.0f) ? (rng / QMAX_A) : 1.0f;
    const float inv   = 1.0f / scale;
    const float zpf   = rintf(-xmin * inv);
    const int   zp    = (int)zpf;

#pragma unroll
    for (int i = 0; i < 4; ++i) {
        int q0 = min(255, max(0, (int)rintf(v[i].x * inv) + zp));
        int q1 = min(255, max(0, (int)rintf(v[i].y * inv) + zp));
        int q2 = min(255, max(0, (int)rintf(v[i].z * inv) + zp));
        int q3 = min(255, max(0, (int)rintf(v[i].w * inv) + zp));
        const unsigned packed = (unsigned)q0 | ((unsigned)q1 << 8) |
                                ((unsigned)q2 << 16) | ((unsigned)q3 << 24);
        *(unsigned*)(Xq + (size_t)m * K + i * 1024 + t * 4) = packed;
    }
    if (t == 0) {
        ascale[m] = scale;
        azp[m]    = zpf;
    }
}

#define LDS_STRIDE 80
#define BUF_BYTES  (128 * LDS_STRIDE)

__device__ __forceinline__ void compute_step(const uint8_t* __restrict__ aB,
                                             const uint8_t* __restrict__ bB,
                                             v8i (&acc)[4][2]) {
    v8i bfrag[2];
#pragma unroll
    for (int nt = 0; nt < 2; ++nt) {
        const uint8_t* bp = bB + nt * (16 * LDS_STRIDE);
        const int4 lo = *(const int4*)bp;
        const int4 hi = *(const int4*)(bp + 32);
        v8i b;
        b[0] = lo.x; b[1] = lo.y; b[2] = lo.z; b[3] = lo.w;
        b[4] = hi.x; b[5] = hi.y; b[6] = hi.z; b[7] = hi.w;
        bfrag[nt] = b;
    }
    v8i afrag[4];
#pragma unroll
    for (int mt = 0; mt < 4; ++mt) {
        const uint8_t* ap = aB + mt * (16 * LDS_STRIDE);
        v8i a;
#pragma unroll
        for (int j = 0; j < 4; ++j) {
            const int2 d = *(const int2*)(ap + 8 * j + 16 * (j >> 1));
            a[2 * j]     = d.x;
            a[2 * j + 1] = d.y;
        }
        afrag[mt] = a;
    }
#pragma unroll
    for (int mt = 0; mt < 4; ++mt)
#pragma unroll
        for (int nt = 0; nt < 2; ++nt)
            acc[mt][nt] = __builtin_amdgcn_wmma_i32_16x16x64_iu8(
                 false, afrag[mt],  true, bfrag[nt],
                acc[mt][nt],  false,  false);
}

__global__ __launch_bounds__(256) void gemm_kernel(
    const uint8_t* __restrict__ Xq, const int8_t* __restrict__ Wq,
    const float* __restrict__ ascale, const float* __restrict__ azp,
    const float* __restrict__ wscale, const int* __restrict__ wsum,
    const float* __restrict__ bias, float* __restrict__ Y,
    int N, int K) {
    __shared__ uint8_t smemA[2][BUF_BYTES];
    __shared__ uint8_t smemB[2][BUF_BYTES];

    const int t      = threadIdx.x;
    const int lane   = t & 31;
    const int wave   = t >> 5;
    const int waveM  = wave & 1;
    const int waveN  = wave >> 1;
    const int blockM = blockIdx.y * 128;
    const int blockN = blockIdx.x * 128;
    const int l15    = lane & 15;
    const int lhalf  = lane >> 4;

    const uint32_t ldsA0 = (uint32_t)(uintptr_t)&smemA[0][0];
    const uint32_t ldsB0 = (uint32_t)(uintptr_t)&smemB[0][0];

    const uint8_t* aB0 = &smemA[0][(waveM * 64 + l15) * LDS_STRIDE + lhalf * 16];
    const uint8_t* aB1 = aB0 + BUF_BYTES;
    const uint8_t* bB0 = &smemB[0][(waveN * 32 + l15) * LDS_STRIDE + lhalf * 16];
    const uint8_t* bB1 = bB0 + BUF_BYTES;

    const int row0  = t >> 2;
    const int part0 = (t & 3) * 16;
    const uint8_t* aGlob = Xq + (size_t)(blockM + row0) * K + part0;
    const int8_t*  bGlob = Wq + (size_t)(blockN + row0) * K + part0;
    const uint32_t cLds  = (uint32_t)(row0 * LDS_STRIDE + part0);
    const size_t   half  = (size_t)64 * K;
    const uint32_t halfL = 64 * LDS_STRIDE;

    const v8i zero = {0, 0, 0, 0, 0, 0, 0, 0};
    v8i acc[4][2];
#pragma unroll
    for (int mt = 0; mt < 4; ++mt)
#pragma unroll
        for (int nt = 0; nt < 2; ++nt)
            acc[mt][nt] = zero;

    const int KSTEPS = K >> 6;

#define STAGE(kb, bufsel)                                                     \
    do {                                                                      \
        const uint32_t bo  = (uint32_t)(bufsel) * (uint32_t)BUF_BYTES;        \
        uint64_t ga0 = (uint64_t)(uintptr_t)(aGlob + (kb));                   \
        uint64_t ga1 = (uint64_t)(uintptr_t)(aGlob + (kb) + half);            \
        uint64_t gb0 = (uint64_t)(uintptr_t)(bGlob + (kb));                   \
        uint64_t gb1 = (uint64_t)(uintptr_t)(bGlob + (kb) + half);            \
        uint32_t la0 = ldsA0 + bo + cLds, la1 = la0 + halfL;                  \
        uint32_t lb0 = ldsB0 + bo + cLds, lb1 = lb0 + halfL;                  \
        asm volatile("global_load_async_to_lds_b128 %0, %1, off"              \
                     :: "v"(la0), "v"(ga0) : "memory");                       \
        asm volatile("global_load_async_to_lds_b128 %0, %1, off"              \
                     :: "v"(la1), "v"(ga1) : "memory");                       \
        asm volatile("global_load_async_to_lds_b128 %0, %1, off"              \
                     :: "v"(lb0), "v"(gb0) : "memory");                       \
        asm volatile("global_load_async_to_lds_b128 %0, %1, off"              \
                     :: "v"(lb1), "v"(gb1) : "memory");                       \
    } while (0)

#define ROTATE()                                                              \
    do {                                                                      \
        asm volatile("s_wait_asynccnt 0x0" ::: "memory");                     \
        __syncthreads();                                                      \
    } while (0)

    STAGE(0, 0);
    ROTATE();

    for (int kbi = 0; kbi < KSTEPS; kbi += 2) {
        if (kbi + 1 < KSTEPS) STAGE((kbi + 1) << 6, 1);
        compute_step(aB0, bB0, acc);
        ROTATE();
        if (kbi + 1 < KSTEPS) {
            if (kbi + 2 < KSTEPS) STAGE((kbi + 2) << 6, 0);
            compute_step(aB1, bB1, acc);
            ROTATE();
        }
    }
#undef STAGE
#undef ROTATE

    const int wMoff = blockM + waveM * 64;
    const int wNoff = blockN + waveN * 32;
#pragma unroll
    for (int mt = 0; mt < 4; ++mt) {
#pragma unroll
        for (int nt = 0; nt < 2; ++nt) {
            const int   n  = wNoff + nt * 16 + l15;
            const float sw = wscale[n];
            const float rs = (float)wsum[n];
            const float bi = bias[n];
#pragma unroll
            for (int i = 0; i < 8; ++i) {
                const int   m    = wMoff + mt * 16 + lhalf * 8 + i;
                const float sa   = ascale[m];
                const float zp   = azp[m];
                const float vacc = (float)acc[mt][nt][i] - zp * rs;
                Y[(size_t)m * N + n] = vacc * (sa * sw) + bi;
            }
        }
    }
}

extern "C" void kernel_launch(void* const* d_in, const int* in_sizes, int n_in,
                              void* d_out, int out_size, void* d_ws, size_t ws_size,
                              hipStream_t stream) {
    (void)n_in; (void)out_size; (void)ws_size;
    const float* x    = (const float*)d_in[0];
    const float* w    = (const float*)d_in[1];
    const float* bias = (const float*)d_in[2];

    const int N = in_sizes[2];
    const int K = in_sizes[1] / N;
    const int M = in_sizes[0] / K;

    auto a256 = [](size_t v) { return (v + 255) & ~(size_t)255; };
    char*  ws  = (char*)d_ws;
    size_t off = 0;
    uint8_t* Xq  = (uint8_t*)(ws + off); off = a256(off + (size_t)M * K);
    int8_t*  Wq  = (int8_t*)(ws + off);  off = a256(off + (size_t)N * K);
    float*   as  = (float*)(ws + off);   off = a256(off + (size_t)M * 4);
    float*   az  = (float*)(ws + off);   off = a256(off + (size_t)M * 4);
    float*   wsc = (float*)(ws + off);   off = a256(off + (size_t)N * 4);
    int*     wsm = (int*)(ws + off);     off = a256(off + (size_t)N * 4);

    qw_kernel<<<N, 256, 0, stream>>>(w, Wq, wsc, wsm, K);
    qx_kernel<<<M, 256, 0, stream>>>(x, Xq, as, az, K);

    dim3 grid(N / 128, M / 128);
    gemm_kernel<<<grid, 256, 0, stream>>>(Xq, Wq, as, az, wsc, wsm, bias,
                                          (float*)d_out, N, K);
}
